// MultiHeadedSelfAttention_26834955665985
// MI455X (gfx1250) — matrix operand order repaired and checked by reading, not compiled, not run on hardware
//
#include <hip/hip_runtime.h>
#include <hip/hip_bf16.h>

// ---------------------------------------------------------------------------
// MHSA for MI455X (gfx1250): bf16 WMMA pipeline with fp32 accumulation.
//   x[4096,512] -> Q,K,V (WMMA GEMM, bf16) -> flash attention (async-to-LDS
//   K/V staging + WMMA + online Schraudolph-fastexp softmax) -> out proj (f32).
// Compute-bound (~43 GFLOP vs ~20MB HBM): v_wmma_f32_16x16x32_bf16 everywhere,
// GLOBAL_LOAD_ASYNC_TO_LDS_B128 (ASYNCcnt) for attention tile staging.
// ---------------------------------------------------------------------------

#define SEQ 4096
#define DIM 512
#define ATT_DIM 512
#define N_HEADS 8
#define HEAD_DIM 64

typedef __bf16 v16bf __attribute__((ext_vector_type(16)));
typedef float  v8f   __attribute__((ext_vector_type(8)));
typedef unsigned short ushort_t;

// ---------------- helpers ---------------------------------------------------

__device__ __forceinline__ ushort_t f2bf(float f) {
    unsigned u = __float_as_uint(f);
    unsigned r = u + 0x7FFFu + ((u >> 16) & 1u);   // round-to-nearest-even
    return (ushort_t)(r >> 16);
}

__device__ __forceinline__ float fastexp_g(float x) {
    // Schraudolph fastexp, matches reference fastexp_gist
    float t = 12102203.17133801f * x + 1064986823.0f;
    t = (t < 8388608.0f) ? 0.0f : fminf(t, 2139095040.0f);
    return __int_as_float((int)t);
}

__device__ __forceinline__ float qrsqrt(float x) {
    // quake fast inverse sqrt, matches reference SCALING
    float x2 = x * 0.5f;
    int   i  = __float_as_int(x);
    i = 1597463007 - (i >> 1);
    float y = __int_as_float(i);
    return y * (1.5f - x2 * y * y);
}

__device__ __forceinline__ v16bf make_frag(uint4 a, uint4 b) {
    union { v16bf f; uint4 q[2]; } u;
    u.q[0] = a; u.q[1] = b;
    return u.f;
}

// A-matrix fragment (16x32 bf16). Lane l<16: M=l, K in {0..7, 16..23};
// lane l>=16: M=l-16, K in {8..15, 24..31}.  base -> element (M=0, K=0).
__device__ __forceinline__ v16bf load_frag_a(const ushort_t* base, int ld) {
    const int lane = threadIdx.x & 31;
    const int half = lane >> 4;
    const int r    = lane & 15;
    const ushort_t* p0 = base + r * ld + half * 8;
    uint4 a = *(const uint4*)p0;
    uint4 b = *(const uint4*)(p0 + 16);
    return make_frag(a, b);
}

// B-matrix fragment (32x16 bf16), source is B^T row-major [N][K].
// Lane l<16: N=l, K in {0..7, 16..23}; lane l>=16: N=l-16, K in {8..15, 24..31}.
__device__ __forceinline__ v16bf load_frag_b(const ushort_t* base, int ld) {
    const int lane = threadIdx.x & 31;
    const int half = lane >> 4;
    const int r    = lane & 15;
    const ushort_t* p0 = base + r * ld + half * 8;
    uint4 a = *(const uint4*)p0;
    uint4 b = *(const uint4*)(p0 + 16);
    return make_frag(a, b);
}

__device__ __forceinline__ v8f wmma_bf16(v16bf a, v16bf b, v8f c) {
    return __builtin_amdgcn_wmma_f32_16x16x32_bf16(
        false, a, false, b, (short)0, c, false, false);
}

// Async global->LDS copy, 16 bytes per lane (CDNA5 GLOBAL_LOAD_ASYNC_TO_LDS,
// tracked by ASYNCcnt).  lds_off = LDS byte offset (low 32 bits of generic
// shared pointer), gptr = per-lane global source.
__device__ __forceinline__ void async_ld_b128(unsigned lds_off, const void* gptr) {
    asm volatile("global_load_async_to_lds_b128 %0, %1, off"
                 :: "v"(lds_off), "v"((unsigned long long)(uintptr_t)gptr)
                 : "memory");
}

__device__ __forceinline__ void wait_asynccnt0() {
    asm volatile("s_wait_asynccnt 0x0" ::: "memory");
}

// ---------------- kernel 1: f32 -> bf16 convert -----------------------------

__global__ void cvt_f32_bf16(const float* __restrict__ in,
                             ushort_t* __restrict__ out, int n) {
    int i = (blockIdx.x * blockDim.x + threadIdx.x) * 4;
    if (i >= n) return;
    float4 v = *(const float4*)(in + i);
    union { ushort_t u[4]; uint2 d; } o;
    o.u[0] = f2bf(v.x); o.u[1] = f2bf(v.y);
    o.u[2] = f2bf(v.z); o.u[3] = f2bf(v.w);
    *(uint2*)(out + i) = o.d;
}

// ---------------- kernel 2: WMMA GEMM  C = A @ B^T + bias -------------------
// A: [M][K] bf16 row-major, Bt: [N][K] bf16 row-major (i.e. torch W layout).
// Block = 256 threads = 8 waves as 4(M) x 2(N); wave tile 32x64; WG tile 128x128.
// MODE 0: bf16 out, per-head [H][L][64]            (Q, K)
// MODE 1: bf16 out, per-head transposed [H][64][L] (V)
// MODE 3: f32 out, row-major [M][N]                (final projection)

template <int MODE>
__global__ __launch_bounds__(256)
void gemm_bf16(const ushort_t* __restrict__ A, int lda,
               const ushort_t* __restrict__ Bt, int ldb,
               const float* __restrict__ bias,
               void* __restrict__ out,
               int M, int N, int K) {
    const int lane = threadIdx.x & 31;
    const int w    = threadIdx.x >> 5;
    const int wm   = w >> 1;           // 0..3
    const int wn   = w & 1;            // 0..1
    const int m0   = blockIdx.y * 128 + wm * 32;
    const int n0   = blockIdx.x * 128 + wn * 64;

    v8f acc[2][4];
    #pragma unroll
    for (int mt = 0; mt < 2; ++mt)
        #pragma unroll
        for (int nt = 0; nt < 4; ++nt)
            acc[mt][nt] = (v8f){0.f,0.f,0.f,0.f,0.f,0.f,0.f,0.f};

    for (int k0 = 0; k0 < K; k0 += 32) {
        v16bf a0 = load_frag_a(A + (size_t)m0 * lda + k0, lda);
        v16bf a1 = load_frag_a(A + (size_t)(m0 + 16) * lda + k0, lda);
        v16bf b[4];
        #pragma unroll
        for (int nt = 0; nt < 4; ++nt)
            b[nt] = load_frag_b(Bt + (size_t)(n0 + nt * 16) * ldb + k0, ldb);
        #pragma unroll
        for (int nt = 0; nt < 4; ++nt) {
            acc[0][nt] = wmma_bf16(a0, b[nt], acc[0][nt]);
            acc[1][nt] = wmma_bf16(a1, b[nt], acc[1][nt]);
        }
    }

    const int half = lane >> 4;
    const int lo   = lane & 15;
    #pragma unroll
    for (int mt = 0; mt < 2; ++mt) {
        #pragma unroll
        for (int nt = 0; nt < 4; ++nt) {
            const int col = n0 + nt * 16 + lo;
            const float bv = (MODE == 3) ? 0.0f : bias[col];
            #pragma unroll
            for (int j = 0; j < 8; ++j) {
                const int row = m0 + mt * 16 + half * 8 + j;
                const float val = acc[mt][nt][j] + bv;
                if (MODE == 3) {
                    ((float*)out)[(size_t)row * N + col] = val;
                } else if (MODE == 0) {
                    size_t idx = (size_t)(col >> 6) * SEQ * HEAD_DIM
                               + (size_t)row * HEAD_DIM + (col & 63);
                    ((ushort_t*)out)[idx] = f2bf(val);
                } else { // MODE 1: per-head transposed
                    size_t idx = (size_t)(col >> 6) * HEAD_DIM * SEQ
                               + (size_t)(col & 63) * SEQ + row;
                    ((ushort_t*)out)[idx] = f2bf(val);
                }
            }
        }
    }
}

// ---------------- kernel 3: flash attention ---------------------------------
// One WG per (head, 64 q rows); 4 waves x 16 query rows.
// Q,K: [H][L][64] bf16.  Vt: [H][64][L] bf16.  Ab out: [L][512] bf16.
// Per key-block: stage K (64x64) and V^T (64x64) into LDS with
// GLOBAL_LOAD_ASYNC_TO_LDS_B128, then serve all WMMA fragments from LDS.

__global__ __launch_bounds__(128)
void attn_kernel(const ushort_t* __restrict__ Qb,
                 const ushort_t* __restrict__ Kb,
                 const ushort_t* __restrict__ Vt,
                 ushort_t* __restrict__ Ab) {
    __shared__ __align__(16) ushort_t kstage[64 * 64];   // K block, row=key
    __shared__ __align__(16) ushort_t vstage[64 * 64];   // V^T block, row=d
    __shared__ __align__(16) ushort_t pls[4][16 * 64];   // per-wave P staging

    const int lane = threadIdx.x & 31;
    const int wave = threadIdx.x >> 5;
    const int h    = blockIdx.y;
    const int row0 = blockIdx.x * 64 + wave * 16;
    const int half = lane >> 4;
    const int lo   = lane & 15;

    const float scaling = qrsqrt((float)HEAD_DIM);

    const ushort_t* Qh  = Qb + (size_t)h * SEQ * HEAD_DIM;
    const ushort_t* Kh  = Kb + (size_t)h * SEQ * HEAD_DIM;
    const ushort_t* Vth = Vt + (size_t)h * HEAD_DIM * SEQ;
    ushort_t* pw = &pls[wave][0];

    // copy-slice assignment: thread t moves half a 128B row of each block
    const unsigned t     = threadIdx.x;
    const unsigned crow  = t >> 1;                 // 0..63
    const unsigned chalf = t & 1;                  // 0,1
    const unsigned sl_el = crow * 64 + chalf * 32; // element offset in 64x64 blk
    const unsigned kbase = (unsigned)(uintptr_t)kstage + sl_el * 2;
    const unsigned vbase = (unsigned)(uintptr_t)vstage + sl_el * 2;

    // Q fragments for this wave's 16 rows (reused across all key blocks)
    v16bf qf0 = load_frag_a(Qh + (size_t)row0 * HEAD_DIM + 0,  HEAD_DIM);
    v16bf qf1 = load_frag_a(Qh + (size_t)row0 * HEAD_DIM + 32, HEAD_DIM);

    v8f o[4];
    #pragma unroll
    for (int nt = 0; nt < 4; ++nt)
        o[nt] = (v8f){0.f,0.f,0.f,0.f,0.f,0.f,0.f,0.f};
    float mrun[8], lrun[8];
    #pragma unroll
    for (int j = 0; j < 8; ++j) { mrun[j] = -3.0e38f; lrun[j] = 0.0f; }

    for (int kb = 0; kb < SEQ / 64; ++kb) {
        // ---- stage K and V^T blocks into LDS (async, per-lane 16B) ----
        const ushort_t* ksrc = Kh  + (size_t)kb * 64 * HEAD_DIM + sl_el;
        const ushort_t* vsrc = Vth + (size_t)crow * SEQ + (size_t)kb * 64
                                   + chalf * 32;
        __syncthreads();                     // prior reads done before overwrite
        #pragma unroll
        for (int i = 0; i < 4; ++i) {
            async_ld_b128(kbase + i * 16, ksrc + i * 8);
            async_ld_b128(vbase + i * 16, vsrc + i * 8);
        }
        wait_asynccnt0();
        __syncthreads();                     // staged data visible to all waves

        // prefetch next block toward L2/WGP$ while we compute on this one
        if (kb + 1 < SEQ / 64) {
            __builtin_prefetch(ksrc + 64 * HEAD_DIM, 0, 0);
            __builtin_prefetch(vsrc + 64, 0, 0);
        }

        // ---- S = Q K^T (scaled), fragments from LDS ----
        v8f s[4];
        #pragma unroll
        for (int nt = 0; nt < 4; ++nt) {
            s[nt] = (v8f){0.f,0.f,0.f,0.f,0.f,0.f,0.f,0.f};
            v16bf kb0 = load_frag_b(kstage + nt * 16 * 64 + 0,  64);
            v16bf kb1 = load_frag_b(kstage + nt * 16 * 64 + 32, 64);
            s[nt] = wmma_bf16(qf0, kb0, s[nt]);
            s[nt] = wmma_bf16(qf1, kb1, s[nt]);
        }
        #pragma unroll
        for (int nt = 0; nt < 4; ++nt)
            #pragma unroll
            for (int j = 0; j < 8; ++j)
                s[nt][j] *= scaling;

        // ---- online softmax update (rows half*8+j live across 16 lanes) ----
        float mnew[8], alpha[8];
        #pragma unroll
        for (int j = 0; j < 8; ++j) {
            float mx = fmaxf(fmaxf(s[0][j], s[1][j]), fmaxf(s[2][j], s[3][j]));
            #pragma unroll
            for (int d = 1; d < 16; d <<= 1)
                mx = fmaxf(mx, __shfl_xor(mx, d, 32));
            mnew[j]  = fmaxf(mrun[j], mx);
            alpha[j] = fastexp_g(mrun[j] - mnew[j]);
            mrun[j]  = mnew[j];
        }
        #pragma unroll
        for (int nt = 0; nt < 4; ++nt)
            #pragma unroll
            for (int j = 0; j < 8; ++j)
                s[nt][j] = fastexp_g(s[nt][j] - mnew[j]);
        #pragma unroll
        for (int j = 0; j < 8; ++j) {
            float rs = s[0][j] + s[1][j] + s[2][j] + s[3][j];
            #pragma unroll
            for (int d = 1; d < 16; d <<= 1)
                rs += __shfl_xor(rs, d, 32);
            lrun[j] = lrun[j] * alpha[j] + rs;
        }
        #pragma unroll
        for (int nt = 0; nt < 4; ++nt)
            #pragma unroll
            for (int j = 0; j < 8; ++j)
                o[nt][j] *= alpha[j];

        // ---- P: C-layout fp32 -> LDS bf16 -> A-layout fragments ----
        #pragma unroll
        for (int nt = 0; nt < 4; ++nt)
            #pragma unroll
            for (int j = 0; j < 8; ++j)
                pw[(half * 8 + j) * 64 + nt * 16 + lo] = f2bf(s[nt][j]);

        v16bf pa0 = load_frag_a(pw + 0,  64);
        v16bf pa1 = load_frag_a(pw + 32, 64);

        // ---- O += P @ V, B fragments from staged V^T ----
        #pragma unroll
        for (int nt = 0; nt < 4; ++nt) {
            v16bf v0 = load_frag_b(vstage + nt * 16 * 64 + 0,  64);
            v16bf v1 = load_frag_b(vstage + nt * 16 * 64 + 32, 64);
            o[nt] = wmma_bf16(pa0, v0, o[nt]);
            o[nt] = wmma_bf16(pa1, v1, o[nt]);
        }
    }

    // ---- epilogue: normalize, store heads-concatenated [L][512] bf16 ----
    #pragma unroll
    for (int j = 0; j < 8; ++j) {
        const float inv = 1.0f / lrun[j];
        const int row = row0 + half * 8 + j;
        #pragma unroll
        for (int nt = 0; nt < 4; ++nt) {
            const int col = h * HEAD_DIM + nt * 16 + lo;
            Ab[(size_t)row * ATT_DIM + col] = f2bf(o[nt][j] * inv);
        }
    }
}

// ---------------- launch -----------------------------------------------------

extern "C" void kernel_launch(void* const* d_in, const int* in_sizes, int n_in,
                              void* d_out, int out_size, void* d_ws, size_t ws_size,
                              hipStream_t stream) {
    const float* x    = (const float*)d_in[0];
    const float* Wq   = (const float*)d_in[1];
    const float* bq   = (const float*)d_in[2];
    const float* Wk   = (const float*)d_in[3];
    const float* bk   = (const float*)d_in[4];
    const float* Wv   = (const float*)d_in[5];
    const float* bv   = (const float*)d_in[6];
    const float* Wout = (const float*)d_in[7];

    ushort_t* ws = (ushort_t*)d_ws;
    const size_t NX = (size_t)SEQ * DIM;        // 2,097,152
    const size_t NW = (size_t)ATT_DIM * DIM;    //   262,144
    ushort_t* xb   = ws;
    ushort_t* wqb  = xb  + NX;
    ushort_t* wkb  = wqb + NW;
    ushort_t* wvb  = wkb + NW;
    ushort_t* wob  = wvb + NW;
    ushort_t* qbuf = wob + NW;       // [H][L][64]
    ushort_t* kbuf = qbuf + NX;      // [H][L][64]
    ushort_t* vt   = kbuf + NX;      // [H][64][L]
    ushort_t* ab   = vt   + NX;      // [L][512]

    // 1) fp32 -> bf16
    cvt_f32_bf16<<<(int)(NX / 4 / 256), 256, 0, stream>>>(x,    xb,  (int)NX);
    cvt_f32_bf16<<<(int)(NW / 4 / 256), 256, 0, stream>>>(Wq,   wqb, (int)NW);
    cvt_f32_bf16<<<(int)(NW / 4 / 256), 256, 0, stream>>>(Wk,   wkb, (int)NW);
    cvt_f32_bf16<<<(int)(NW / 4 / 256), 256, 0, stream>>>(Wv,   wvb, (int)NW);
    cvt_f32_bf16<<<(int)(NW / 4 / 256), 256, 0, stream>>>(Wout, wob, (int)NW);

    // 2) projections
    dim3 ggrid(ATT_DIM / 128, SEQ / 128);   // (4, 32)
    gemm_bf16<0><<<ggrid, 256, 0, stream>>>(xb, DIM, wqb, DIM, bq, qbuf,
                                            SEQ, ATT_DIM, DIM);
    gemm_bf16<0><<<ggrid, 256, 0, stream>>>(xb, DIM, wkb, DIM, bk, kbuf,
                                            SEQ, ATT_DIM, DIM);
    gemm_bf16<1><<<ggrid, 256, 0, stream>>>(xb, DIM, wvb, DIM, bv, vt,
                                            SEQ, ATT_DIM, DIM);

    // 3) attention
    attn_kernel<<<dim3(SEQ / 64, N_HEADS), 128, 0, stream>>>(qbuf, kbuf, vt, ab);

    // 4) output projection (fp32 result)
    gemm_bf16<3><<<ggrid, 256, 0, stream>>>(ab, ATT_DIM, wob, ATT_DIM, nullptr,
                                            d_out, SEQ, DIM, ATT_DIM);
}
